// RobustGCN_52192442581284
// MI455X (gfx1250) — compile-verified, matrix operand order checked by reading, not run on hardware
//
#include <hip/hip_runtime.h>
#include <hip/hip_bf16.h>
#include <math.h>

#define NN 100000
#define FF 512
#define HH 128
#define CC 8
#define EE 1600000
#define GAMMA_C 1.0f
#define EPS_C 1e-8f
#define ASTRIDE 40          // padded LDS row stride in bf16 elems (80 B): conflict-free, 16B aligned
#define ROWB (ASTRIDE * 2)  // 80 bytes

typedef __attribute__((ext_vector_type(16))) __bf16 v16bf;
typedef __attribute__((ext_vector_type(8)))  float  v8f;
typedef __attribute__((ext_vector_type(4)))  __bf16 bf16x4;
typedef __attribute__((ext_vector_type(4)))  unsigned int u32x4;
typedef __attribute__((ext_vector_type(4)))  float f32x4;

__device__ __forceinline__ float elu1f(float x) {
    return x > 0.f ? x : (__expf(x) - 1.f);
}

// per-lane async DMA: 16 bytes global -> LDS, tracked by ASYNCcnt
__device__ __forceinline__ void async_copy_b128(unsigned lds_off, const void* gptr) {
    asm volatile("global_load_async_to_lds_b128 %0, %1, off"
                 :: "v"(lds_off), "v"(gptr)
                 : "memory");
}
__device__ __forceinline__ void wait_asynccnt0() {
    asm volatile("s_wait_asynccnt 0x0" ::: "memory");
}
__device__ __forceinline__ unsigned lds_addr32(const void* p) {
    return (unsigned)(unsigned long long)p;   // flat->LDS uses addr[31:0]
}

// ---------------- kernel 0a: weights -> bf16, transposed [h][k] ----------------
__global__ __launch_bounds__(256) void cvt_w_kernel(const float* __restrict__ Wm1,
                                                    const float* __restrict__ Ws1,
                                                    __bf16* __restrict__ WmT,
                                                    __bf16* __restrict__ WsT) {
    int i = blockIdx.x * 256 + threadIdx.x;   // FF*HH = 65536
    if (i < FF * HH) {
        int k = i >> 7;        // 0..511
        int h = i & 127;       // 0..127
        WmT[(size_t)h * FF + k] = (__bf16)Wm1[i];
        WsT[(size_t)h * FF + k] = (__bf16)Ws1[i];
    }
}

// ---------------- kernel 0b: X -> bf16 row-major ----------------
__global__ __launch_bounds__(256) void cvt_x_kernel(const float* __restrict__ X,
                                                    __bf16* __restrict__ Xb) {
    long long i = (long long)blockIdx.x * 256 + threadIdx.x;   // over NN*FF/4 float4s
    if (i < (long long)NN * FF / 4) {
        f32x4 v = *(const f32x4*)&X[i * 4];
        bf16x4 b;
        b.x = (__bf16)v.x; b.y = (__bf16)v.y; b.z = (__bf16)v.z; b.w = (__bf16)v.w;
        *(bf16x4*)&Xb[i * 4] = b;
    }
}

// ---------------- kernel Z: zero accumulators ----------------
__global__ __launch_bounds__(256) void zero_kernel(float* __restrict__ miu_h,
                                                   float* __restrict__ sigma_h) {
    int i = blockIdx.x * 256 + threadIdx.x;   // NN*HH
    if (i < NN * HH) { miu_h[i] = 0.f; sigma_h[i] = 0.f; }
}

// ---------------- kernel 1: fused dual GEMM (X@Wm1, X@Ws1), async-LDS + WMMA bf16 ----
// block tile: 128 rows x 64 cols of H, K=512 in steps of 32.
// 8 waves: wm 0..3 (32 rows), wn 0..1 (32 cols); 2x2 wmma tiles per wave, both weights.
__global__ __launch_bounds__(256)
void gemm1_kernel(const __bf16* __restrict__ Xb,
                  const __bf16* __restrict__ WmT,   // [HH][FF] bf16
                  const __bf16* __restrict__ WsT,   // [HH][FF] bf16
                  __bf16* __restrict__ P,           // miu1*att1      (N x H)
                  __bf16* __restrict__ Q)           // sigma1*att1^2  (N x H)
{
    __shared__ __bf16 As [128 * ASTRIDE];
    __shared__ __bf16 Btm[ 64 * ASTRIDE];   // Wm tile [col][k]
    __shared__ __bf16 Bts[ 64 * ASTRIDE];   // Ws tile [col][k]

    const int tid  = threadIdx.x;
    const int wave = tid >> 5;
    const int lane = tid & 31;
    const int half = lane >> 4;     // K-group select per 16-bit WMMA layout
    const int l16  = lane & 15;
    const int wm   = wave >> 1;     // 0..3
    const int wn   = wave & 1;      // 0..1
    const int rowBase = blockIdx.x * 128;
    const int hb      = blockIdx.y * 64;

    const unsigned asBase  = lds_addr32(&As[0]);
    const unsigned btmBase = lds_addr32(&Btm[0]);
    const unsigned btsBase = lds_addr32(&Bts[0]);

    // per-thread staging coordinates (constant across K loop)
    const int arow0 = (tid)        >> 2;          // A chunk 0 row (0..63)
    const int arow1 = (tid + 256)  >> 2;          // A chunk 1 row (64..127)
    const int ac4   = tid & 3;                    // 16B chunk within 64B row
    const int brow  = tid >> 2;                   // B row (0..63)

    v8f accm[2][2], accs[2][2];
    #pragma unroll
    for (int i = 0; i < 2; ++i)
        #pragma unroll
        for (int j = 0; j < 2; ++j) {
            accm[i][j] = (v8f){0.f,0.f,0.f,0.f,0.f,0.f,0.f,0.f};
            accs[i][j] = (v8f){0.f,0.f,0.f,0.f,0.f,0.f,0.f,0.f};
        }

    for (int k0 = 0; k0 < FF; k0 += 32) {
        __syncthreads();   // all fragment reads from previous step complete

        // ---- A tile: 128 rows x 64B, two 16B chunks per thread (async DMA)
        {
            int g0 = rowBase + arow0;
            if (g0 < NN)
                async_copy_b128(asBase + arow0 * ROWB + ac4 * 16,
                                &Xb[(size_t)g0 * FF + k0 + ac4 * 8]);
            int g1 = rowBase + arow1;
            if (g1 < NN)
                async_copy_b128(asBase + arow1 * ROWB + ac4 * 16,
                                &Xb[(size_t)g1 * FF + k0 + ac4 * 8]);
        }
        // ---- B tiles: 64 rows x 64B each, one 16B chunk per thread per tile
        {
            const size_t goff = (size_t)(hb + brow) * FF + k0 + ac4 * 8;
            async_copy_b128(btmBase + brow * ROWB + ac4 * 16, &WmT[goff]);
            async_copy_b128(btsBase + brow * ROWB + ac4 * 16, &WsT[goff]);
            if (k0 + 32 < FF)
                __builtin_prefetch(&WmT[goff + 32], 0, 1);   // global_prefetch_b8
        }
        wait_asynccnt0();
        __syncthreads();   // all waves' async data visible in LDS

        // ---- fragments (16-bit layout: lanes<16 hold K 0..7 & 16..23; lanes>=16 hold 8..15 & 24..31)
        v16bf afrag[2], bfm[2], bfs[2];
        #pragma unroll
        for (int mt = 0; mt < 2; ++mt) {
            const __bf16* base = &As[(wm * 32 + mt * 16 + l16) * ASTRIDE];
            union { u32x4 q[2]; v16bf v; } f;
            f.q[0] = *(const u32x4*)(base + 8 * half);
            f.q[1] = *(const u32x4*)(base + 16 + 8 * half);
            afrag[mt] = f.v;
        }
        #pragma unroll
        for (int nt = 0; nt < 2; ++nt) {
            const __bf16* bm = &Btm[(wn * 32 + nt * 16 + l16) * ASTRIDE];
            const __bf16* bs = &Bts[(wn * 32 + nt * 16 + l16) * ASTRIDE];
            union { u32x4 q[2]; v16bf v; } fm, fs;
            fm.q[0] = *(const u32x4*)(bm + 8 * half);
            fm.q[1] = *(const u32x4*)(bm + 16 + 8 * half);
            fs.q[0] = *(const u32x4*)(bs + 8 * half);
            fs.q[1] = *(const u32x4*)(bs + 16 + 8 * half);
            bfm[nt] = fm.v;
            bfs[nt] = fs.v;
        }
        #pragma unroll
        for (int mt = 0; mt < 2; ++mt)
            #pragma unroll
            for (int nt = 0; nt < 2; ++nt) {
                accm[mt][nt] = __builtin_amdgcn_wmma_f32_16x16x32_bf16(
                    false, afrag[mt], false, bfm[nt], (short)0, accm[mt][nt], false, false);
                accs[mt][nt] = __builtin_amdgcn_wmma_f32_16x16x32_bf16(
                    false, afrag[mt], false, bfs[nt], (short)0, accs[mt][nt], false, false);
            }
    }

    // ---- epilogue: elu/relu/att, store P = miu*att, Q = sig*att^2 (bf16)
    #pragma unroll
    for (int mt = 0; mt < 2; ++mt)
        #pragma unroll
        for (int nt = 0; nt < 2; ++nt) {
            int col = hb + wn * 32 + nt * 16 + l16;
            #pragma unroll
            for (int r = 0; r < 8; ++r) {
                int m = rowBase + wm * 32 + mt * 16 + r + 8 * half;
                if (m < NN) {
                    float mu  = accm[mt][nt][r];
                    float sg  = accs[mt][nt][r];
                    float mue = elu1f(mu);
                    float sgr = sg > 0.f ? sg : 0.f;
                    float att = __expf(-GAMMA_C * sgr);
                    P[(size_t)m * HH + col] = (__bf16)(mue * att);
                    Q[(size_t)m * HH + col] = (__bf16)(sgr * att * att);
                }
            }
        }
}

// ---------------- kernel 2: edge aggregation #1 (one wave per edge, 4 h per lane) ----
__global__ __launch_bounds__(256)
void agg1_kernel(const int* __restrict__ esrc, const int* __restrict__ edst,
                 const float* __restrict__ w1, const float* __restrict__ w2,
                 const __bf16* __restrict__ P, const __bf16* __restrict__ Q,
                 float* __restrict__ miu_h, float* __restrict__ sigma_h)
{
    int t    = blockIdx.x * 256 + threadIdx.x;
    int e    = t >> 5;
    int lane = t & 31;
    if (e >= EE) return;
    int s = esrc[e], d = edst[e];
    float a = w1[e], b = w2[e];
    int h = lane * 4;
    bf16x4 p = *(const bf16x4*)&P[(size_t)s * HH + h];
    bf16x4 q = *(const bf16x4*)&Q[(size_t)s * HH + h];
    #pragma unroll
    for (int i = 0; i < 4; ++i) {
        atomicAdd(&miu_h  [(size_t)d * HH + h + i], a * (float)p[i]);
        atomicAdd(&sigma_h[(size_t)d * HH + h + i], b * (float)q[i]);
    }
}

// ---------------- kernel 3: layer-2 dense (H->C) + activations ----------------
__global__ __launch_bounds__(256)
void layer2_kernel(const float* __restrict__ miu_h, const float* __restrict__ sigma_h,
                   const float* __restrict__ Wm2, const float* __restrict__ Ws2,
                   float* __restrict__ R,        // miu2*att2 (N x C)
                   float* __restrict__ sig2)     // sigma2    (N x C)
{
    __shared__ float wm[HH * CC], ws[HH * CC];
    for (int i = threadIdx.x; i < HH * CC; i += 256) { wm[i] = Wm2[i]; ws[i] = Ws2[i]; }
    __syncthreads();
    int n = blockIdx.x * 256 + threadIdx.x;
    if (n >= NN) return;
    float am[CC] = {0}, as_[CC] = {0};
    for (int k = 0; k < HH; ++k) {
        float mh = miu_h  [(size_t)n * HH + k];
        float sh = sigma_h[(size_t)n * HH + k];
        #pragma unroll
        for (int c = 0; c < CC; ++c) {
            am[c]  += mh * wm[k * CC + c];
            as_[c] += sh * ws[k * CC + c];
        }
    }
    #pragma unroll
    for (int c = 0; c < CC; ++c) {
        float miu2 = elu1f(am[c]);
        float s2   = as_[c] > 0.f ? as_[c] : 0.f;
        float att  = __expf(-GAMMA_C * s2);
        R   [(size_t)n * CC + c] = miu2 * att;
        sig2[(size_t)n * CC + c] = s2;
    }
}

// ---------------- kernel 4: init d_out = noise * sqrt(sigma2 + eps) ----------------
__global__ __launch_bounds__(256)
void initout_kernel(const float* __restrict__ noise, const float* __restrict__ sig2,
                    float* __restrict__ out)
{
    int i = blockIdx.x * 256 + threadIdx.x;
    if (i < NN * CC) out[i] = noise[i] * sqrtf(sig2[i] + EPS_C);
}

// ---------------- kernel 5: edge aggregation #2 into d_out (8 lanes per edge) ----
__global__ __launch_bounds__(256)
void agg2_kernel(const int* __restrict__ esrc, const int* __restrict__ edst,
                 const float* __restrict__ w1, const float* __restrict__ R,
                 float* __restrict__ out)
{
    int t = blockIdx.x * 256 + threadIdx.x;
    int e = t >> 3;
    int c = t & 7;
    if (e >= EE) return;
    int s = esrc[e], d = edst[e];
    atomicAdd(&out[(size_t)d * CC + c], w1[e] * R[(size_t)s * CC + c]);
}

// ---------------- host launcher ----------------
extern "C" void kernel_launch(void* const* d_in, const int* in_sizes, int n_in,
                              void* d_out, int out_size, void* d_ws, size_t ws_size,
                              hipStream_t stream) {
    const float* features = (const float*)d_in[0];
    const int*   edge_src = (const int*)  d_in[1];
    const int*   edge_dst = (const int*)  d_in[2];
    const float* w1       = (const float*)d_in[3];
    const float* w2       = (const float*)d_in[4];
    const float* Wm1      = (const float*)d_in[5];
    const float* Ws1      = (const float*)d_in[6];
    const float* Wm2      = (const float*)d_in[7];
    const float* Ws2      = (const float*)d_in[8];
    const float* noise    = (const float*)d_in[9];
    float* out = (float*)d_out;

    auto alignup = [](size_t x) { return (x + 255) & ~(size_t)255; };
    char* ws = (char*)d_ws;
    size_t off = 0;
    __bf16* WmT    = (__bf16*)(ws + off); off = alignup(off + (size_t)FF * HH * 2);
    __bf16* WsT    = (__bf16*)(ws + off); off = alignup(off + (size_t)FF * HH * 2);
    __bf16* Xb     = (__bf16*)(ws + off); off = alignup(off + (size_t)NN * FF * 2);
    __bf16* P      = (__bf16*)(ws + off); off = alignup(off + (size_t)NN * HH * 2);
    __bf16* Q      = (__bf16*)(ws + off); off = alignup(off + (size_t)NN * HH * 2);
    float*  miu_h  = (float*) (ws + off); off = alignup(off + (size_t)NN * HH * 4);
    float*  sigma_h= (float*) (ws + off); off = alignup(off + (size_t)NN * HH * 4);
    float*  R      = (float*) (ws + off); off = alignup(off + (size_t)NN * CC * 4);
    float*  sig2   = (float*) (ws + off); off = alignup(off + (size_t)NN * CC * 4);
    (void)ws_size; (void)in_sizes; (void)n_in; (void)out_size;

    cvt_w_kernel<<<(FF * HH + 255) / 256, 256, 0, stream>>>(Wm1, Ws1, WmT, WsT);
    cvt_x_kernel<<<(int)(((size_t)NN * FF / 4 + 255) / 256), 256, 0, stream>>>(features, Xb);
    zero_kernel<<<(NN * HH + 255) / 256, 256, 0, stream>>>(miu_h, sigma_h);
    {
        dim3 grid((NN + 127) / 128, HH / 64, 1);
        gemm1_kernel<<<grid, 256, 0, stream>>>(Xb, WmT, WsT, P, Q);
    }
    agg1_kernel<<<(EE * 32) / 256, 256, 0, stream>>>(edge_src, edge_dst, w1, w2,
                                                     P, Q, miu_h, sigma_h);
    layer2_kernel<<<(NN + 255) / 256, 256, 0, stream>>>(miu_h, sigma_h, Wm2, Ws2, R, sig2);
    initout_kernel<<<(NN * CC + 255) / 256, 256, 0, stream>>>(noise, sig2, out);
    agg2_kernel<<<(EE * 8) / 256, 256, 0, stream>>>(edge_src, edge_dst, w1, R, out);
}
